// QLlamaMLP_4475355922947
// MI455X (gfx1250) — matrix operand order repaired and checked by reading, not compiled, not run on hardware
//
#include <hip/hip_runtime.h>

// ---------------------------------------------------------------------------
// QLlamaMLP for MI455X (gfx1250, wave32).
// Pipeline: global-amax -> fp8 block QDQ acts (q1/q2 outlier residuals)
// -> int4 block QDQ weights -> f16 WMMA GEMMs (v_wmma_f32_16x16x32_f16,
//    64x64 per wave, async global->LDS staging via GLOBAL_LOAD_ASYNC_TO_LDS)
// -> SiLU*up + amax -> mid QDQ -> final WMMA GEMM * gm into d_out.
// ---------------------------------------------------------------------------

typedef __attribute__((ext_vector_type(16))) _Float16 v16h;
typedef __attribute__((ext_vector_type(8)))  _Float16 v8h;
typedef __attribute__((ext_vector_type(8)))  float    v8f;

#define GSCALE_DIV 2688.0f   // 448 * 6

// ----------------------------- quantizers ----------------------------------

// RNE quantize-dequantize to OCP e4m3 grid; |x| must be <= 448 on entry.
__device__ __forceinline__ float qdq_e4m3(float x) {
    float ax = fabsf(x);
    if (ax == 0.0f) return 0.0f;
    int e;
    (void)frexpf(ax, &e);                    // ax = m * 2^e, m in [0.5, 1)
    int ue = (e >= -5) ? (e - 4) : -9;       // normal ulp 2^(e-4); subnormal 2^-9
    float ulp = ldexpf(1.0f, ue);
    float q = nearbyintf(x / ulp) * ulp;     // round-to-nearest-even
    return fminf(fmaxf(q, -448.0f), 448.0f);
}

__device__ __forceinline__ void fp8_block_qdq(const float* v, float* q) {
    float amax = 0.0f;
#pragma unroll
    for (int t = 0; t < 16; ++t) amax = fmaxf(amax, fabsf(v[t]));
    float s = fmaxf(amax / 448.0f, 1e-8f);
#pragma unroll
    for (int t = 0; t < 16; ++t) q[t] = qdq_e4m3(v[t] / s) * s;
}

// ------------------------------- kernels ------------------------------------

__global__ void init_scal_kernel(float* s) { s[0] = 0.0f; s[1] = 0.0f; }

__global__ void absmax_kernel(const float* __restrict__ x, size_t n, float* out) {
    __shared__ float red[256];
    size_t i = (size_t)blockIdx.x * blockDim.x + threadIdx.x;
    size_t stride = (size_t)gridDim.x * blockDim.x;
    float m = 0.0f;
    for (; i < n; i += stride) m = fmaxf(m, fabsf(x[i]));
    red[threadIdx.x] = m;
    __syncthreads();
    for (int s2 = 128; s2 > 0; s2 >>= 1) {
        if ((int)threadIdx.x < s2)
            red[threadIdx.x] = fmaxf(red[threadIdx.x], red[threadIdx.x + s2]);
        __syncthreads();
    }
    if (threadIdx.x == 0)
        atomicMax((unsigned int*)out, __float_as_uint(red[0]));  // vals >= 0
}

// One thread per (row, 16-block) of the output activation matrix A (M x CA).
// Columns [0,n): normal fp8 QDQ; [n,C): q1 of outliers; [C,CA): q2 residual.
__global__ void quant_act_kernel(const float* __restrict__ X, int M, int C, int n,
                                 int CA, const int* __restrict__ ri,
                                 const float* __restrict__ amax_ptr,
                                 _Float16* __restrict__ A) {
    const int nb = CA / 16;
    long idx = (long)blockIdx.x * blockDim.x + threadIdx.x;
    if (idx >= (long)M * nb) return;
    int m  = (int)(idx / nb);
    int b  = (int)(idx % nb);
    int j0 = b * 16;
    int js = (j0 < C) ? j0 : (j0 - 128);      // q2 region re-reads outlier cols
    float g = fmaxf(amax_ptr[0] / GSCALE_DIV, 1e-8f);
    const float* xr = X + (size_t)m * C;
    float v[16], q[16];
#pragma unroll
    for (int t = 0; t < 16; ++t) v[t] = xr[ri[js + t]] / g;
    fp8_block_qdq(v, q);                      // normal, or q1 for outliers
    if (j0 >= C) {                            // q2 = qdq(outl - q1)
        float r2[16];
#pragma unroll
        for (int t = 0; t < 16; ++t) r2[t] = v[t] - q[t];
        fp8_block_qdq(r2, q);
    }
    _Float16* out = A + (size_t)m * CA + j0;
#pragma unroll
    for (int t = 0; t < 16; ++t) out[t] = (_Float16)q[t];
}

// One thread per (row, 16-block) of WQ (N x CW); int4 block QDQ with the
// outlier columns duplicated at [C, CW).
__global__ void quant_w_kernel(const float* __restrict__ W, int N, int C, int CW,
                               const int* __restrict__ ri,
                               _Float16* __restrict__ WQ) {
    const int nb = CW / 16;
    long idx = (long)blockIdx.x * blockDim.x + threadIdx.x;
    if (idx >= (long)N * nb) return;
    int row = (int)(idx / nb);
    int b   = (int)(idx % nb);
    int j0  = b * 16;
    int js  = (j0 < C) ? j0 : (j0 - 128);
    const float* wr = W + (size_t)row * C;
    float v[16];
    float amax = 0.0f;
#pragma unroll
    for (int t = 0; t < 16; ++t) { v[t] = wr[ri[js + t]]; amax = fmaxf(amax, fabsf(v[t])); }
    float s = fmaxf(amax / 7.0f, 1e-8f);
    _Float16* out = WQ + (size_t)row * CW + j0;
#pragma unroll
    for (int t = 0; t < 16; ++t) {
        float q = rintf(fminf(fmaxf(v[t] / s, -8.0f), 7.0f)) * s;  // RNE = jnp.round
        out[t] = (_Float16)q;
    }
}

// h = silu(gate*g) * (up*g), in place over gate; also reduces amax(|h|).
__global__ void silu_mul_amax_kernel(const float* gate, const float* up,
                                     const float* __restrict__ amax_x,
                                     float* h, size_t n, float* amax_h) {
    __shared__ float red[256];
    float g = fmaxf(amax_x[0] / GSCALE_DIV, 1e-8f);
    size_t i = (size_t)blockIdx.x * blockDim.x + threadIdx.x;
    size_t stride = (size_t)gridDim.x * blockDim.x;
    float m = 0.0f;
    for (; i < n; i += stride) {
        float gt = gate[i] * g;
        float u  = up[i] * g;
        float hv = (gt / (1.0f + expf(-gt))) * u;
        h[i] = hv;
        m = fmaxf(m, fabsf(hv));
    }
    red[threadIdx.x] = m;
    __syncthreads();
    for (int s2 = 128; s2 > 0; s2 >>= 1) {
        if ((int)threadIdx.x < s2)
            red[threadIdx.x] = fmaxf(red[threadIdx.x], red[threadIdx.x + s2]);
        __syncthreads();
    }
    if (threadIdx.x == 0)
        atomicMax((unsigned int*)amax_h, __float_as_uint(red[0]));
}

// ------------------------- f16 WMMA GEMM (C = A @ B^T) ----------------------
// A: M x K row-major f16.  B: N x K row-major f16.  C: M x N f32.
// Workgroup: 256 threads = 8 waves, tile 128(M) x 256(N), K-step 32.
// Staging: GLOBAL_LOAD_ASYNC_TO_LDS_B128 (ASYNCcnt) into double-buffered LDS
// (stride 40 halves to spread banks); each wave drains its own ASYNCcnt with
// s_wait_asynccnt before the barrier publishes the tile.
// Wave (wm in 0..1, wn in 0..3) owns 64x64 via 4x4 v_wmma_f32_16x16x32_f16.
__global__ __launch_bounds__(256)
void gemm_f16_wmma_kernel(const _Float16* __restrict__ A,
                          const _Float16* __restrict__ B,
                          float* __restrict__ Cmat,
                          int M, int N, int K,
                          const float* __restrict__ scale_amax) {
    __shared__ _Float16 sA[2][128][40];   // 2 x 10240 B
    __shared__ _Float16 sB[2][256][40];   // 2 x 20480 B

    const int tid  = threadIdx.x;
    const int lane = tid & 31;
    const int wv   = tid >> 5;
    const int wm   = wv & 1;        // 0..1  -> 64-row slab
    const int wn   = wv >> 1;       // 0..3  -> 64-col slab
    const int hf   = lane >> 4;     // half-wave select
    const int rr   = lane & 15;     // row/col within 16-tile
    const int bm   = blockIdx.y * 128;
    const int bn   = blockIdx.x * 256;

    // Async staging map: A tile 128x32 halves -> thread t covers row t/2,
    // 16 halves at col (t&1)*16 (2 x b128). B tile 256x32 halves -> thread t
    // covers full row t (4 x b128). INST_OFFSET applies to both addresses.
    const int arow = tid >> 1;
    const int acol = (tid & 1) * 16;
    const _Float16* gA = A + (size_t)(bm + arow) * K + acol;
    const _Float16* gB = B + (size_t)(bn + tid) * K;

    v8f acc[4][4];
    v8f zero = {};
#pragma unroll
    for (int i = 0; i < 4; ++i)
#pragma unroll
        for (int j = 0; j < 2; ++j) { acc[i][j] = zero; acc[i][j + 2] = zero; }

    // Prologue: stage K-tile 0 into buffer 0.
    {
        unsigned la = (unsigned)(size_t)&sA[0][arow][acol];
        unsigned long long ga = (unsigned long long)(size_t)gA;
        asm volatile(
            "global_load_async_to_lds_b128 %0, %1, off\n\t"
            "global_load_async_to_lds_b128 %0, %1, off offset:16"
            :: "v"(la), "v"(ga) : "memory");
        unsigned lb = (unsigned)(size_t)&sB[0][tid][0];
        unsigned long long gb = (unsigned long long)(size_t)gB;
        asm volatile(
            "global_load_async_to_lds_b128 %0, %1, off\n\t"
            "global_load_async_to_lds_b128 %0, %1, off offset:16\n\t"
            "global_load_async_to_lds_b128 %0, %1, off offset:32\n\t"
            "global_load_async_to_lds_b128 %0, %1, off offset:48"
            :: "v"(lb), "v"(gb) : "memory");
        asm volatile("s_wait_asynccnt 0x0" ::: "memory");
    }
    __syncthreads();

    const int nK = K >> 5;
    for (int kb = 0; kb < nK; ++kb) {
        const int cur = kb & 1;
        const int nxt = cur ^ 1;
        const bool more = (kb + 1) < nK;
        if (more) {  // async-stage next K tile into the other buffer
            unsigned la = (unsigned)(size_t)&sA[nxt][arow][acol];
            unsigned long long ga =
                (unsigned long long)(size_t)(gA + (size_t)(kb + 1) * 32);
            asm volatile(
                "global_load_async_to_lds_b128 %0, %1, off\n\t"
                "global_load_async_to_lds_b128 %0, %1, off offset:16"
                :: "v"(la), "v"(ga) : "memory");
            unsigned lb = (unsigned)(size_t)&sB[nxt][tid][0];
            unsigned long long gb =
                (unsigned long long)(size_t)(gB + (size_t)(kb + 1) * 32);
            asm volatile(
                "global_load_async_to_lds_b128 %0, %1, off\n\t"
                "global_load_async_to_lds_b128 %0, %1, off offset:16\n\t"
                "global_load_async_to_lds_b128 %0, %1, off offset:32\n\t"
                "global_load_async_to_lds_b128 %0, %1, off offset:48"
                :: "v"(lb), "v"(gb) : "memory");
        }

        // A fragment: lane half hf holds K {hf*8..hf*8+7} U {16+hf*8..}
        v16h af[4];
#pragma unroll
        for (int i = 0; i < 4; ++i) {
            const _Float16* p = &sA[cur][wm * 64 + i * 16 + rr][hf * 8];
            v8h lo = *(const v8h*)(p);
            v8h hi = *(const v8h*)(p + 16);
            af[i] = __builtin_shufflevector(lo, hi, 0, 1, 2, 3, 4, 5, 6, 7,
                                            8, 9, 10, 11, 12, 13, 14, 15);
        }
        // B fragment: lane half hf holds K {hf*8..hf*8+7} U {16+hf*8..}
        v16h bf[4];
#pragma unroll
        for (int j = 0; j < 4; ++j) {
            const _Float16* p = &sB[cur][wn * 64 + j * 16 + rr][hf * 8];
            v8h lo = *(const v8h*)(p);
            v8h hi = *(const v8h*)(p + 16);
            bf[j] = __builtin_shufflevector(lo, hi, 0, 1, 2, 3, 4, 5, 6, 7,
                                            8, 9, 10, 11, 12, 13, 14, 15);
        }
#pragma unroll
        for (int i = 0; i < 4; ++i)
#pragma unroll
            for (int j = 0; j < 4; ++j)
                acc[i][j] = __builtin_amdgcn_wmma_f32_16x16x32_f16(
                    false, af[i], false, bf[j], (short)0, acc[i][j], false, false);

        if (more)  // drain own async loads, then barrier publishes the tile
            asm volatile("s_wait_asynccnt 0x0" ::: "memory");
        __syncthreads();
    }

    float scale = 1.0f;
    if (scale_amax) scale = fmaxf(scale_amax[0] / GSCALE_DIV, 1e-8f);

    // C/D layout: lane rr = column, vgpr v = row v + 8*hf within 16x16 tile.
#pragma unroll
    for (int i = 0; i < 4; ++i)
#pragma unroll
        for (int j = 0; j < 4; ++j) {
            float* pc = Cmat + (size_t)(bm + wm * 64 + i * 16 + hf * 8) * N
                             + (bn + wn * 64 + j * 16 + rr);
#pragma unroll
            for (int v = 0; v < 8; ++v) pc[(size_t)v * N] = acc[i][j][v] * scale;
        }
}

// ------------------------------- launcher -----------------------------------

extern "C" void kernel_launch(void* const* d_in, const int* in_sizes, int n_in,
                              void* d_out, int out_size, void* d_ws, size_t ws_size,
                              hipStream_t stream) {
    (void)in_sizes; (void)n_in; (void)out_size; (void)ws_size;

    const float* x   = (const float*)d_in[0];
    const float* wgp = (const float*)d_in[1];
    const float* wup = (const float*)d_in[2];
    const float* wdp = (const float*)d_in[3];
    const int* ri_in  = (const int*)d_in[4];
    const int* ri_mid = (const int*)d_in[5];
    float* y = (float*)d_out;

    const int M = 4096;                 // B*L
    const int H = 4096;
    const int I = 11008;
    const int KA = H + 128;             // 4224 = 132*32
    const int KM = I + 128;             // 11136 = 348*32

    char* ws   = (char*)d_ws;
    float* scal = (float*)ws;           // [0]=amax(x), [1]=amax(h)
    _Float16* Aq  = (_Float16*)(ws + 256);
    size_t oWg   = 256 + (size_t)M * KA * 2;
    _Float16* Wgq = (_Float16*)(ws + oWg);
    size_t oWu   = oWg + (size_t)I * KA * 2;
    _Float16* Wuq = (_Float16*)(ws + oWu);
    size_t oGate = oWu + (size_t)I * KA * 2;
    float* gate  = (float*)(ws + oGate);
    size_t oUp   = oGate + (size_t)M * I * 4;
    float* up    = (float*)(ws + oUp);
    _Float16* Amq = Wgq;                // Wg consumed before Am is written
    _Float16* Wdq = Wuq;                // Wu consumed before Wd is written
    float* h = gate;                    // SiLU in place

    init_scal_kernel<<<1, 1, 0, stream>>>(scal);
    absmax_kernel<<<2048, 256, 0, stream>>>(x, (size_t)M * H, &scal[0]);

    {
        long total = (long)M * (KA / 16);
        int blocks = (int)((total + 255) / 256);
        quant_act_kernel<<<blocks, 256, 0, stream>>>(x, M, H, H - 128, KA, ri_in,
                                                     &scal[0], Aq);
    }
    {
        long total = (long)I * (KA / 16);
        int blocks = (int)((total + 255) / 256);
        quant_w_kernel<<<blocks, 256, 0, stream>>>(wgp, I, H, KA, ri_in, Wgq);
        quant_w_kernel<<<blocks, 256, 0, stream>>>(wup, I, H, KA, ri_in, Wuq);
    }

    gemm_f16_wmma_kernel<<<dim3(I / 256, M / 128), 256, 0, stream>>>(
        Aq, Wgq, gate, M, I, KA, nullptr);
    gemm_f16_wmma_kernel<<<dim3(I / 256, M / 128), 256, 0, stream>>>(
        Aq, Wuq, up, M, I, KA, nullptr);

    silu_mul_amax_kernel<<<4096, 256, 0, stream>>>(gate, up, &scal[0], h,
                                                   (size_t)M * I, &scal[1]);

    {
        long total = (long)M * (KM / 16);
        int blocks = (int)((total + 255) / 256);
        quant_act_kernel<<<blocks, 256, 0, stream>>>(h, M, I, I - 128, KM, ri_mid,
                                                     &scal[1], Amq);
    }
    {
        long total = (long)H * (KM / 16);
        int blocks = (int)((total + 255) / 256);
        quant_w_kernel<<<blocks, 256, 0, stream>>>(wdp, H, I, KM, ri_mid, Wdq);
    }

    gemm_f16_wmma_kernel<<<dim3(H / 256, M / 128), 256, 0, stream>>>(
        Amq, Wdq, y, M, H, KM, &scal[1]);
}
